// voxel_fea_50422916055478
// MI455X (gfx1250) — matrix operand order repaired and checked by reading, not compiled, not run on hardware
//
#include <hip/hip_runtime.h>
#include <hip/hip_bf16.h>

// ---- problem constants (match reference setup_inputs) ----
#define GRID_E 480
#define NBATCH 4
#define NPER   100000
#define BN     400000           // NBATCH*NPER, == 3125*128 exactly
#define BINS   (NBATCH*GRID_E*GRID_E)  // 921600 == 900*1024 exactly
#define FDIM   3
#define H1     128
#define H2     256
#define DOUT   64

typedef __attribute__((ext_vector_type(16))) _Float16 v16h;
typedef __attribute__((ext_vector_type(8)))  _Float16 v8h;
typedef __attribute__((ext_vector_type(8)))  float    v8f;

union F16x16 { v16h v; v8h h[2]; };

// ---- workspace layout (bytes) ----
#define WS_SCAN   0u                       // u32[BINS]      (flags, then inclusive scan)
#define WS_BSUMS  3686400u                 // u32[1024]
#define WS_LIN    3690496u                 // u32[BN]
#define WS_W2T    5290496u                 // f16[H2*H1]   W2 transposed [n][k]
#define WS_W3T    5356032u                 // f16[DOUT*H2] W3 transposed [d][k]

__device__ __forceinline__ void atomic_max_f32(float* addr, float v) {
    asm volatile("global_atomic_max_num_f32 %0, %1, off" :: "v"(addr), "v"(v) : "memory");
}

// ------------------------------------------------------------------
__global__ void k_zero_scan(unsigned* scan) {
    unsigned i = blockIdx.x * 256 + threadIdx.x;
    scan[i] = 0u;                           // grid sized exactly BINS/256
}

__global__ void k_mark(const int* xy, unsigned* scan, unsigned* lin) {
    unsigned p = blockIdx.x * 256 + threadIdx.x;
    if (p >= BN) return;
    int b = p / NPER;
    int x = xy[2 * p + 0];
    int y = xy[2 * p + 1];
    unsigned l = (unsigned)((b * GRID_E + x) * GRID_E + y);
    lin[p]  = l;
    scan[l] = 1u;                           // races write identical value
}

// per-block inclusive scan of 1024 flags (256 thr x 4), block total -> bsums
__global__ void k_scan_block(unsigned* scan, unsigned* bsums) {
    __shared__ unsigned s[256];
    int t = threadIdx.x;
    unsigned base = blockIdx.x * 1024u + (unsigned)t * 4u;
    unsigned a0 = scan[base + 0], a1 = scan[base + 1];
    unsigned a2 = scan[base + 2], a3 = scan[base + 3];
    a1 += a0; a2 += a1; a3 += a2;
    s[t] = a3;
    __syncthreads();
    #pragma unroll
    for (int off = 1; off < 256; off <<= 1) {
        unsigned v = (t >= off) ? s[t - off] : 0u;
        __syncthreads();
        s[t] += v;
        __syncthreads();
    }
    unsigned pre = (t > 0) ? s[t - 1] : 0u;
    scan[base + 0] = a0 + pre;
    scan[base + 1] = a1 + pre;
    scan[base + 2] = a2 + pre;
    scan[base + 3] = a3 + pre;
    if (t == 255) bsums[blockIdx.x] = s[255];
}

__global__ void k_scan_sums(unsigned* bsums) {   // single block, 1024 threads
    __shared__ unsigned s[1024];
    int t = threadIdx.x;
    s[t] = (t < 900) ? bsums[t] : 0u;
    __syncthreads();
    for (int off = 1; off < 1024; off <<= 1) {
        unsigned v = (t >= off) ? s[t - off] : 0u;
        __syncthreads();
        s[t] += v;
        __syncthreads();
    }
    if (t < 900) bsums[t] = s[t];
}

__global__ void k_add_offsets(unsigned* scan, const unsigned* bsums) {
    if (blockIdx.x == 0) return;
    unsigned add  = bsums[blockIdx.x - 1];
    unsigned base = blockIdx.x * 1024u + threadIdx.x * 4u;
    scan[base + 0] += add; scan[base + 1] += add;
    scan[base + 2] += add; scan[base + 3] += add;
}

// decode present bins -> unq rows (sorted order == linear bin order)
__global__ void k_write_unq(const unsigned* scan, float* out) {
    unsigned bin = blockIdx.x * 256 + threadIdx.x;   // grid == BINS/256
    unsigned v    = scan[bin];
    unsigned prev = (bin > 0) ? scan[bin - 1] : 0u;
    if (v != prev) {
        unsigned seg = prev;                 // == v-1
        unsigned b = bin / (GRID_E * GRID_E);
        unsigned r = bin % (GRID_E * GRID_E);
        out[seg * 3 + 0] = (float)b;
        out[seg * 3 + 1] = (float)(r / GRID_E);
        out[seg * 3 + 2] = (float)(r % GRID_E);
    }
}

__global__ void k_init_pooled(float* pooled, int count) {
    int i = blockIdx.x * 256 + threadIdx.x;
    if (i < count) pooled[i] = -__builtin_inff();
}

// W2 (H1 x H2, row-major [k][n]) -> W2T f16 [n][k]; W3 (H2 x DOUT) -> W3T [d][k]
__global__ void k_prep_weights(const float* W2, const float* W3,
                               _Float16* W2T, _Float16* W3T) {
    int i = blockIdx.x * 256 + threadIdx.x;
    if (i < H2 * H1) {
        int n = i / H1, k = i % H1;
        W2T[n * H1 + k] = (_Float16)W2[k * H2 + n];
    } else {
        int t = i - H2 * H1;                 // t < DOUT*H2 by grid sizing
        int d = t / H2, k = t % H2;
        W3T[d * H2 + k] = (_Float16)W3[k * DOUT + d];
    }
}

// ------------------------------------------------------------------
// Fused MLP + segment-max scatter. 256 threads (8 waves) per 128-point tile.
__global__ __launch_bounds__(256)
void k_mlp_pool(const float* __restrict__ pt,
                const float* __restrict__ W1, const float* __restrict__ b1,
                const float* __restrict__ b2, const float* __restrict__ b3,
                const _Float16* __restrict__ W2T, const _Float16* __restrict__ W3T,
                const unsigned* __restrict__ scan, const unsigned* __restrict__ lin,
                float* __restrict__ pooled) {
    extern __shared__ char smem[];
    _Float16* h1  = (_Float16*)smem;                    // 128*128 f16 = 32 KB
    _Float16* h2  = (_Float16*)(smem + 32768);          // 128*256 f16 = 64 KB
    float*    w1s = (float*)(smem + 32768 + 65536);     // 384
    float*    b1s = w1s + FDIM * H1;                    // 128
    float*    b2s = b1s + H1;                           // 256
    float*    b3s = b2s + H2;                           // 64

    const int tid  = threadIdx.x;
    const long base = (long)blockIdx.x * 128;

    for (int i = tid; i < FDIM * H1; i += 256) w1s[i] = W1[i];
    for (int i = tid; i < H1;        i += 256) b1s[i] = b1[i];
    for (int i = tid; i < H2;        i += 256) b2s[i] = b2[i];
    for (int i = tid; i < DOUT;      i += 256) b3s[i] = b3[i];
    __syncthreads();

    // ---- layer 1 (K=3, VALU): 2 threads per point, 64 outputs each ----
    {
        int m    = tid >> 1;
        int j0   = (tid & 1) * 64;
        long p   = base + m;                            // 3125*128 == BN exactly
        float x0 = pt[p * 3 + 0], x1 = pt[p * 3 + 1], x2 = pt[p * 3 + 2];
        #pragma unroll 8
        for (int j = 0; j < 64; ++j) {
            int jj  = j0 + j;
            float a = b1s[jj] + x0 * w1s[jj] + x1 * w1s[H1 + jj] + x2 * w1s[2 * H1 + jj];
            h1[m * H1 + jj] = (_Float16)(a > 0.f ? a : 0.f);
        }
    }
    __syncthreads();

    const int wv   = tid >> 5;        // wave id == M-tile (0..7)
    const int lane = tid & 31;
    const int c    = lane & 15;       // col / row-within-tile
    const int g    = lane >> 4;       // K-half select

    // ---- layer 2: h2 = relu(h1 @ W2 + b2) via v_wmma_f32_16x16x32_f16 ----
    {
        // A fragments for this wave's 16-row strip, all 4 K-steps (reused 16x)
        F16x16 a2[4];
        const _Float16* arow = h1 + (wv * 16 + c) * H1;
        #pragma unroll
        for (int kk = 0; kk < 4; ++kk) {
            const _Float16* p0 = arow + kk * 32 + g * 16;
            a2[kk].h[0] = *(const v8h*)(p0);
            a2[kk].h[1] = *(const v8h*)(p0 + 8);
        }
        #pragma unroll 4
        for (int nt = 0; nt < 16; ++nt) {
            v8f acc = {};
            const _Float16* bb = W2T + (nt * 16 + c) * H1 + g * 16;
            #pragma unroll
            for (int kk = 0; kk < 4; ++kk) {
                v16h bf = *(const v16h*)(bb + kk * 32);
                acc = __builtin_amdgcn_wmma_f32_16x16x32_f16(
                          false, a2[kk].v, false, bf, (short)0, acc, false, false);
            }
            const int n = nt * 16 + c;
            float bias  = b2s[n];
            #pragma unroll
            for (int j = 0; j < 8; ++j) {
                float v = acc[j] + bias;
                v = v > 0.f ? v : 0.f;
                h2[(wv * 16 + j + g * 8) * H2 + n] = (_Float16)v;
            }
        }
    }
    __syncthreads();

    // ---- layer 3: feat = h2 @ W3 + b3, then atomic segment-max scatter ----
    {
        F16x16 a3[8];
        const _Float16* arow = h2 + (wv * 16 + c) * H2;
        #pragma unroll
        for (int kk = 0; kk < 8; ++kk) {
            const _Float16* p0 = arow + kk * 32 + g * 16;
            a3[kk].h[0] = *(const v8h*)(p0);
            a3[kk].h[1] = *(const v8h*)(p0 + 8);
        }
        // segment ids for this lane's 8 output rows (independent of nt)
        int segj[8];
        #pragma unroll
        for (int j = 0; j < 8; ++j) {
            long p  = base + wv * 16 + j + g * 8;
            segj[j] = (int)(scan[lin[p]] - 1u);
        }
        #pragma unroll
        for (int nt = 0; nt < 4; ++nt) {
            v8f acc = {};
            const _Float16* bb = W3T + (nt * 16 + c) * H2 + g * 16;
            #pragma unroll
            for (int kk = 0; kk < 8; ++kk) {
                v16h bf = *(const v16h*)(bb + kk * 32);
                acc = __builtin_amdgcn_wmma_f32_16x16x32_f16(
                          false, a3[kk].v, false, bf, (short)0, acc, false, false);
            }
            const int n = nt * 16 + c;
            float bias  = b3s[n];
            #pragma unroll
            for (int j = 0; j < 8; ++j) {
                atomic_max_f32(pooled + (long)segj[j] * DOUT + n, acc[j] + bias);
            }
        }
    }
}

// ------------------------------------------------------------------
extern "C" void kernel_launch(void* const* d_in, const int* in_sizes, int n_in,
                              void* d_out, int out_size, void* d_ws, size_t ws_size,
                              hipStream_t stream) {
    const float* pt_fea = (const float*)d_in[0];
    const int*   xy_ind = (const int*)d_in[1];
    const float* W1 = (const float*)d_in[2];
    const float* b1 = (const float*)d_in[3];
    const float* W2 = (const float*)d_in[4];
    const float* b2 = (const float*)d_in[5];
    const float* W3 = (const float*)d_in[6];
    const float* b3 = (const float*)d_in[7];

    char* ws = (char*)d_ws;
    unsigned*  scan  = (unsigned*)(ws + WS_SCAN);
    unsigned*  bsums = (unsigned*)(ws + WS_BSUMS);
    unsigned*  lin   = (unsigned*)(ws + WS_LIN);
    _Float16*  W2T   = (_Float16*)(ws + WS_W2T);
    _Float16*  W3T   = (_Float16*)(ws + WS_W3T);

    float* out = (float*)d_out;
    const int M = out_size / (3 + DOUT);       // unq rows == segments
    float* pooled = out + (long)M * 3;

    k_prep_weights<<<(H2 * H1 + DOUT * H2) / 256, 256, 0, stream>>>(W2, W3, W2T, W3T);
    k_zero_scan  <<<BINS / 256, 256, 0, stream>>>(scan);
    k_mark       <<<(BN + 255) / 256, 256, 0, stream>>>(xy_ind, scan, lin);
    k_scan_block <<<BINS / 1024, 256, 0, stream>>>(scan, bsums);
    k_scan_sums  <<<1, 1024, 0, stream>>>(bsums);
    k_add_offsets<<<BINS / 1024, 256, 0, stream>>>(scan, bsums);
    k_write_unq  <<<BINS / 256, 256, 0, stream>>>(scan, out);
    k_init_pooled<<<(M * DOUT + 255) / 256, 256, 0, stream>>>(pooled, M * DOUT);

    const int smem_bytes = 32768 + 65536 + (FDIM * H1 + H1 + H2 + DOUT) * 4;
    k_mlp_pool<<<BN / 128, 256, smem_bytes, stream>>>(
        pt_fea, W1, b1, b2, b3, W2T, W3T, scan, lin, pooled);
}
